// GraphUnet_57543971832577
// MI455X (gfx1250) — matrix operand order repaired and checked by reading, not compiled, not run on hardware
//
#include <hip/hip_runtime.h>

typedef __attribute__((ext_vector_type(16))) _Float16 v16h;
typedef __attribute__((ext_vector_type(2)))  _Float16 v2h;
typedef __attribute__((ext_vector_type(8)))  float    v8f;

#define DHP 80                    // padded per-head feature dim
#define XPH (3072 * 80)           // per-head stride in xp workspace
#define NEGV -1.0e9f

// A-operand (16x32 f16) fragment order chosen here (contiguous; B uses same): register element i of lane
// (M=lane&15, hi=lane>>4) holds K = i + hi*16.
// Inverse: K=kk maps to slot hi=kk>>4, i=kk&15.
__device__ __forceinline__ int afrag_hi(int kk) { return kk >> 4; }
__device__ __forceinline__ int afrag_i(int kk)  { return kk & 15; }

// ---------------------------------------------------------------------------
// Projection GEMM: xp[h] = X(MxK) @ W[h](KxNd) + b[h], stored MxDHP (pad cols
// [Nd,DHP) zeroed).  Block tile 64x32, 8 waves (4 row-tiles x 2 col-tiles).
// LDS tiles are stored in WMMA fragment order so operand loads are b128s.
// ---------------------------------------------------------------------------
__global__ __launch_bounds__(256) void k_proj(
    const float* __restrict__ X, int lda,
    const float* __restrict__ Wt, int wstride, int ldb,
    const float* __restrict__ bias, int bstride,
    float* __restrict__ xp, int M, int K, int Nd)
{
  __shared__ _Float16 As2[64][2][16] __attribute__((aligned(64)));  // frag-order A
  __shared__ _Float16 BsT[32][32]    __attribute__((aligned(64)));  // [col][k]
  const int tid  = threadIdx.x;
  const int lane = tid & 31;
  const int w    = tid >> 5;
  const int wr = w >> 1, wc = w & 1;
  const int hi = lane >> 4;
  const int mr = lane & 15;
  const int h  = blockIdx.z;
  const float* Wh = Wt + (size_t)h * wstride;
  const float* bh = bias + (size_t)h * bstride;
  float* out = xp + (size_t)h * XPH;
  const int row0 = blockIdx.x * 64;
  const int col0 = blockIdx.y * 32;
  v8f acc = {};
  for (int k0 = 0; k0 < K; k0 += 32) {
    const bool fullA = (row0 + 64 <= M) && (k0 + 32 <= K);
    if (fullA) {
      // 64 rows x 32 k : each thread two float4 loads, packed v2h stores
      #pragma unroll
      for (int it = 0; it < 2; ++it) {
        int e = tid + it * 256;            // 0..511
        int r = e >> 3, kc = (e & 7) * 4;  // kc multiple of 4 -> same hi slot
        const float4 fv = *(const float4*)(X + (size_t)(row0 + r) * lda + k0 + kc);
        int hs = afrag_hi(kc), ib = afrag_i(kc);
        v2h p0; p0[0] = (_Float16)fv.x; p0[1] = (_Float16)fv.y;
        v2h p1; p1[0] = (_Float16)fv.z; p1[1] = (_Float16)fv.w;
        *(v2h*)&As2[r][hs][ib]     = p0;
        *(v2h*)&As2[r][hs][ib + 2] = p1;
      }
    } else {
      #pragma unroll
      for (int i = 0; i < 8; ++i) {
        int e = tid + i * 256;
        int r = e >> 5, kk = e & 31;
        int gr = row0 + r, gk = k0 + kk;
        float fv = (gr < M && gk < K) ? X[(size_t)gr * lda + gk] : 0.f;
        As2[r][afrag_hi(kk)][afrag_i(kk)] = (_Float16)fv;
      }
    }
    {
      // B: 32 k x 32 cols, store transposed [col][k]; clamp-select (no branch)
      const bool fullK = (k0 + 32 <= K);
      #pragma unroll
      for (int i = 0; i < 4; ++i) {
        int e = tid + i * 256;
        int r = e >> 5, n = e & 31;        // r = k index, n = col
        int gk = k0 + r, gn = col0 + n;
        int gks = (gk < K) ? gk : (K - 1);
        int gns = (gn < Nd) ? gn : 0;
        float fv = Wh[(size_t)gks * ldb + gns];
        bool ok = (fullK || gk < K) && (gn < Nd);
        BsT[n][r] = (_Float16)(ok ? fv : 0.f);
      }
    }
    __syncthreads();
    const v16h a = *(const v16h*)As2[wr * 16 + mr][hi];
    const v16h b = *(const v16h*)&BsT[wc * 16 + mr][hi * 16];
    acc = __builtin_amdgcn_wmma_f32_16x16x32_f16(false, a, false, b,
                                                 (short)0, acc, false, false);
    __syncthreads();
  }
  const int col = col0 + wc * 16 + mr;
  if (col < DHP) {
    #pragma unroll
    for (int v = 0; v < 8; ++v) {
      int row = row0 + wr * 16 + v + 8 * hi;
      if (row < M)
        out[(size_t)row * DHP + col] = (col < Nd) ? (acc[v] + bh[col]) : 0.f;
    }
  }
}

// s1[h,n] = xp[h,n,:].phi1[h],  s2[h,n] = xp[h,n,:].phi2[h]
__global__ void k_s1s2(const float* __restrict__ xp, const float* __restrict__ phi,
                       int dh, int N, float* __restrict__ s1, float* __restrict__ s2)
{
  int i = blockIdx.x * blockDim.x + threadIdx.x;
  if (i >= 4 * N) return;
  int h = i / N, n = i % N;
  const float* row = xp + (size_t)h * XPH + (size_t)n * DHP;
  const float* p1 = phi + (size_t)h * 2 * dh;
  const float* p2 = p1 + dh;
  float a = 0.f, b = 0.f;
  for (int c = 0; c < dh; ++c) { a += row[c] * p1[c]; b += row[c] * p2[c]; }
  s1[h * 3072 + n] = a;
  s2[h * 3072 + n] = b;
}

// ---------------------------------------------------------------------------
// Fused flash-style GAT attention. Per wave {head h, 16 p-rows}; 32-wide q
// tiles; P built in registers directly in A-fragment layout; xp tile staged
// transposed so B-fragments are contiguous v16h; adjacency tile staged with
// CDNA5 async global->LDS DMA when fully aligned.
// ---------------------------------------------------------------------------
__global__ __launch_bounds__(256) void k_attn(
    const float* __restrict__ Adj, int N,
    const float* __restrict__ xp,
    const float* __restrict__ s1, const float* __restrict__ s2,
    float* __restrict__ Xout, int ldout, int dh,
    const float* __restrict__ resid, const float* __restrict__ skip)
{
  __shared__ _Float16 xptT[4][80][32] __attribute__((aligned(64)));  // [h][col][k]
  __shared__ float At[2][16][32] __attribute__((aligned(128)));      // contiguous rows
  __shared__ float s2t[4][32];
  const int tid  = threadIdx.x;
  const int lane = tid & 31;
  const int w    = tid >> 5;
  const int h    = w & 3;               // head
  const int rb   = w >> 2;              // row-block within block
  const int hi   = lane >> 4;
  const int mr   = lane & 15;
  const int p0 = blockIdx.x * 32 + rb * 16;
  const int p  = p0 + mr;
  const float s1p = (p < N) ? s1[h * 3072 + p] : 0.f;
  const bool rowsFull = (blockIdx.x * 32 + 32 <= N);
  float m_run = NEGV, l_run = 0.f;
  v8f acc[5] = {{}, {}, {}, {}, {}};

  for (int q0 = 0; q0 < N; q0 += 32) {
    const bool qFull = (q0 + 32 <= N);
    if (q0 + 32 < N) {                    // prefetch next adjacency tile
      int pr = blockIdx.x * 32 + (tid >> 3);
      if (pr >= N) pr = N - 1;
      __builtin_prefetch(&Adj[(size_t)pr * N + q0 + 32], 0, 1);
    }
    // ---- stage xp tile (transposed) ----
    if (qFull) {
      // 1280 units: (head, col-group of 4, k-pair); 2 b128 loads + 4 b32 stores
      #pragma unroll
      for (int it = 0; it < 5; ++it) {
        int u = tid + it * 256;
        int hh = u / 320, rem = u % 320;
        int cg = rem >> 4, kp = rem & 15;
        int c4 = cg * 4, k = kp * 2;
        const float* base = xp + (size_t)hh * XPH + (size_t)(q0 + k) * DHP + c4;
        const float4 x0 = *(const float4*)base;
        const float4 x1 = *(const float4*)(base + DHP);
        v2h pk;
        pk[0] = (_Float16)x0.x; pk[1] = (_Float16)x1.x; *(v2h*)&xptT[hh][c4 + 0][k] = pk;
        pk[0] = (_Float16)x0.y; pk[1] = (_Float16)x1.y; *(v2h*)&xptT[hh][c4 + 1][k] = pk;
        pk[0] = (_Float16)x0.z; pk[1] = (_Float16)x1.z; *(v2h*)&xptT[hh][c4 + 2][k] = pk;
        pk[0] = (_Float16)x0.w; pk[1] = (_Float16)x1.w; *(v2h*)&xptT[hh][c4 + 3][k] = pk;
      }
    } else {
      for (int e = tid; e < 4 * 32 * 80; e += 256) {
        int hh = e / 2560, rem = e % 2560, k = rem / 80, c = rem % 80;
        int q = q0 + k;
        float fv = (q < N) ? xp[(size_t)hh * XPH + (size_t)q * DHP + c] : 0.f;
        xptT[hh][c][k] = (_Float16)fv;
      }
    }
    // ---- stage adjacency tiles ----
    if (rowsFull && qFull && ((N & 3) == 0)) {
      // CDNA5 async DMA: one b128 per lane, 256 lanes -> 4KB tile
      unsigned lds = (unsigned)(size_t)&At[0][0][0] + (unsigned)(tid * 16);
      const float* g = Adj + (size_t)(blockIdx.x * 32 + (tid >> 3)) * N
                           + q0 + (tid & 7) * 4;
      asm volatile("global_load_async_to_lds_b128 %0, %1, off"
                   :: "v"(lds), "v"((unsigned long long)(size_t)g) : "memory");
    } else if (rowsFull && qFull) {
      #pragma unroll
      for (int i = 0; i < 4; ++i) {
        int e = tid + i * 256;
        int rbi = e >> 9, rem = e & 511, m = rem >> 5, k = rem & 31;
        int pr = blockIdx.x * 32 + rbi * 16 + m;
        At[rbi][m][k] = Adj[(size_t)pr * N + q0 + k];
      }
    } else {
      #pragma unroll
      for (int i = 0; i < 4; ++i) {
        int e = tid + i * 256;
        int rbi = e >> 9, rem = e & 511, m = rem >> 5, k = rem & 31;
        int pr = blockIdx.x * 32 + rbi * 16 + m, q = q0 + k;
        At[rbi][m][k] = (pr < N && q < N) ? Adj[(size_t)pr * N + q] : 0.f;
      }
    }
    if (tid < 128) {                      // stage s2
      int hh = tid >> 5, k = tid & 31;
      int q = q0 + k;
      s2t[hh][k] = (q < N) ? s2[hh * 3072 + q] : 0.f;
    }
    asm volatile("s_wait_asynccnt 0x0" ::: "memory");
    __syncthreads();

    // ---- scores in A-fragment order ----
    float pv[16];
    #pragma unroll
    for (int i = 0; i < 16; ++i) {
      int k = i + hi * 16;                      // fragment element -> K
      int q = q0 + k;
      float s = s1p + s2t[h][k];
      s = (s > 0.f) ? s : 0.2f * s;             // leaky relu 0.2
      bool ok = (p < N) && (q < N) &&
                ((At[rb][mr][k] + ((p == q) ? 1.f : 0.f)) > 0.f);
      pv[i] = ok ? s : NEGV;
    }
    float rm = pv[0];
    #pragma unroll
    for (int i = 1; i < 16; ++i) rm = fmaxf(rm, pv[i]);
    rm = fmaxf(rm, __shfl_xor(rm, 16, 32));     // combine K-halves
    float newm  = fmaxf(m_run, rm);
    float scale = __expf(m_run - newm);
    float ev[16], rs = 0.f;
    #pragma unroll
    for (int i = 0; i < 16; ++i) { ev[i] = __expf(pv[i] - newm); rs += ev[i]; }
    rs += __shfl_xor(rs, 16, 32);
    l_run = l_run * scale + rs;
    m_run = newm;
    v16h af;
    #pragma unroll
    for (int i = 0; i < 16; ++i) af[i] = (_Float16)ev[i];
    #pragma unroll
    for (int v = 0; v < 8; ++v) {               // rescale accumulators
      float f = __shfl(scale, v + 8 * hi, 32);
      #pragma unroll
      for (int t = 0; t < 5; ++t) acc[t][v] *= f;
    }
    #pragma unroll
    for (int t = 0; t < 5; ++t) {
      const v16h bf = *(const v16h*)&xptT[h][t * 16 + mr][hi * 16];
      acc[t] = __builtin_amdgcn_wmma_f32_16x16x32_f16(false, af, false, bf,
                                                      (short)0, acc[t], false, false);
    }
    __syncthreads();
  }

  float linv[8];
  #pragma unroll
  for (int v = 0; v < 8; ++v)
    linv[v] = 1.f / __shfl(l_run, v + 8 * hi, 32);
  #pragma unroll
  for (int t = 0; t < 5; ++t) {
    int c = t * 16 + mr;
    if (c >= dh) continue;
    #pragma unroll
    for (int v = 0; v < 8; ++v) {
      int row = p0 + v + 8 * hi;
      if (row < N) {
        float val = acc[t][v] * linv[v];
        size_t off = (size_t)row * ldout + (size_t)(h * dh + c);
        if (resid) val += resid[off];
        if (skip)  val += skip[off];
        Xout[off] = val;
      }
    }
  }
}

// ---------------------------------------------------------------------------
// Pooling machinery
// ---------------------------------------------------------------------------
__global__ void k_pool_scores(const float* __restrict__ X, int N, int C,
                              const float* __restrict__ pw,
                              const float* __restrict__ pb,
                              float* __restrict__ sc)
{
  int n = blockIdx.x * blockDim.x + threadIdx.x;
  if (n >= N) return;
  float z = pb[0];
  for (int c = 0; c < C; ++c) z += X[(size_t)n * C + c] * pw[c];
  sc[n] = 1.f / (1.f + __expf(-z * 0.01f));
}

__global__ __launch_bounds__(1024) void k_topk(const float* __restrict__ sc,
                                               int N, int k,
                                               int* __restrict__ idx,
                                               float* __restrict__ val)
{
  __shared__ float key[4096];
  __shared__ int   ki[4096];
  const int tid = threadIdx.x;
  for (int i = tid; i < 4096; i += 1024) {
    key[i] = (i < N) ? sc[i] : -3.0e38f;
    ki[i]  = i;
  }
  __syncthreads();
  for (int ksz = 2; ksz <= 4096; ksz <<= 1) {
    for (int j = ksz >> 1; j > 0; j >>= 1) {
      for (int i = tid; i < 4096; i += 1024) {
        int ixj = i ^ j;
        if (ixj > i) {
          bool desc = ((i & ksz) == 0);
          float a = key[i], b = key[ixj];
          int ia = ki[i], ib = ki[ixj];
          bool aFirst = (a > b) || (a == b && ia < ib);
          if (desc ? !aFirst : aFirst) {
            key[i] = b; key[ixj] = a; ki[i] = ib; ki[ixj] = ia;
          }
        }
      }
      __syncthreads();
    }
  }
  for (int i = tid; i < k; i += 1024) { idx[i] = ki[i]; val[i] = key[i]; }
}

__global__ void k_gather_rows(const float* __restrict__ Xo,
                              const int* __restrict__ idx,
                              const float* __restrict__ val,
                              int k, int C, float* __restrict__ Xn)
{
  int i = blockIdx.x * blockDim.x + threadIdx.x;
  if (i >= k * C) return;
  int r = i / C, c = i % C;
  Xn[i] = Xo[(size_t)idx[r] * C + c] * val[r];
}

__global__ void k_adj_gather(const float* __restrict__ Ao, int No,
                             const int* __restrict__ idx, int k,
                             float* __restrict__ An)
{
  int j = blockIdx.x * 16 + threadIdx.x;
  int i = blockIdx.y * 16 + threadIdx.y;
  if (i < k && j < k) An[(size_t)i * k + j] = Ao[(size_t)idx[i] * No + idx[j]];
}

__global__ void k_zero(float* __restrict__ p, int n)
{
  int i = blockIdx.x * blockDim.x + threadIdx.x;
  if (i < n) p[i] = 0.f;
}

__global__ void k_scatter_rows(float* __restrict__ Xf,
                               const float* __restrict__ Xs,
                               const int* __restrict__ idx, int k, int C)
{
  int i = blockIdx.x * blockDim.x + threadIdx.x;
  if (i >= k * C) return;
  int r = i / C, c = i % C;
  Xf[(size_t)idx[r] * C + c] = Xs[i];
}

__global__ void k_concat(const float* __restrict__ A, const float* __restrict__ B,
                         int N, int Ca, int Cb, float* __restrict__ D)
{
  int i = blockIdx.x * blockDim.x + threadIdx.x;
  int Ct = Ca + Cb;
  if (i >= N * Ct) return;
  int r = i / Ct, c = i % Ct;
  D[i] = (c < Ca) ? A[(size_t)r * Ca + c] : B[(size_t)r * Cb + (c - Ca)];
}

__global__ void k_copy_cols(const float* __restrict__ S, int lds, int N, int C,
                            float* __restrict__ D)
{
  int i = blockIdx.x * blockDim.x + threadIdx.x;
  if (i >= N * C) return;
  int r = i / C, c = i % C;
  D[i] = S[(size_t)r * lds + c];
}

// ---------------------------------------------------------------------------
extern "C" void kernel_launch(void* const* d_in, const int* in_sizes, int n_in,
                              void* d_out, int out_size, void* d_ws, size_t ws_size,
                              hipStream_t stream)
{
  (void)in_sizes; (void)n_in; (void)out_size; (void)ws_size;
  const float* A0        = (const float*)d_in[0];
  const float* Xin0      = (const float*)d_in[1];
  const float* start_w   = (const float*)d_in[2];
  const float* start_b   = (const float*)d_in[3];
  const float* start_phi = (const float*)d_in[4];
  const float* down_w    = (const float*)d_in[5];
  const float* down_b    = (const float*)d_in[6];
  const float* down_phi  = (const float*)d_in[7];
  const float* pool_w    = (const float*)d_in[8];
  const float* pool_b    = (const float*)d_in[9];
  const float* bottom_w  = (const float*)d_in[10];
  const float* bottom_b  = (const float*)d_in[11];
  const float* bottom_phi= (const float*)d_in[12];
  const float* up_w      = (const float*)d_in[13];
  const float* up_b      = (const float*)d_in[14];
  const float* up_phi    = (const float*)d_in[15];
  const float* end_w     = (const float*)d_in[16];
  const float* end_b     = (const float*)d_in[17];
  const float* end_phi   = (const float*)d_in[18];
  float* out = (float*)d_out;

  const int N0 = 3072, IN = 512, DIM = 300, H = 4, DH = 75, ODIM = 268, DHE = 67;
  const int K1 = 2457, K2 = 1474;   // int(0.8*3072), int(0.6*2457)
  const size_t DW = (size_t)H * DIM * DH, DB = (size_t)H * DH, DP = (size_t)H * 2 * DH;

  float* W = (float*)d_ws;
  size_t o = 0;
  auto alloc = [&](size_t n) { float* p = W + o; o += n; return p; };
  float* xp   = alloc((size_t)4 * 3072 * 80);
  float* s1   = alloc(4 * 3072);
  float* s2   = alloc(4 * 3072);
  float* Borg = alloc((size_t)3072 * 300);
  float* B0   = alloc((size_t)3072 * 300);
  float* B1   = alloc((size_t)3072 * 300);
  float* B2   = alloc((size_t)3072 * 300);
  float* Bd0  = alloc((size_t)3072 * 300);
  float* Bd1  = alloc((size_t)3072 * 300);
  float* Bcat = alloc((size_t)3072 * 600);
  float* A1   = alloc((size_t)K1 * K1);
  float* A2   = alloc((size_t)K2 * K2);
  float* sc   = alloc(3072);
  float* val0 = alloc(3072);
  float* val1 = alloc(3072);
  int* idx0 = (int*)alloc(3072);
  int* idx1 = (int*)alloc(3072);

  auto gat = [&](const float* Adj, int N, const float* Xi, int din,
                 const float* w, const float* b, const float* phi, int dh,
                 float* Xo, int ldout, const float* resid, const float* skip) {
    dim3 gp((N + 63) / 64, 3, 4);
    k_proj<<<gp, 256, 0, stream>>>(Xi, din, w, din * dh, dh, b, dh, xp, N, din, dh);
    int t = 4 * N;
    k_s1s2<<<(t + 255) / 256, 256, 0, stream>>>(xp, phi, dh, N, s1, s2);
    k_attn<<<(N + 31) / 32, 256, 0, stream>>>(Adj, N, xp, s1, s2, Xo, ldout, dh,
                                              resid, skip);
  };

  // start GAT
  gat(A0, N0, Xin0, IN, start_w, start_b, start_phi, DH, Borg, DIM, nullptr, nullptr);
  // down 0 (+residual) -> Bd0, then pool to K1
  gat(A0, N0, Borg, DIM, down_w, down_b, down_phi, DH, Bd0, DIM, Borg, nullptr);
  k_pool_scores<<<(N0 + 255) / 256, 256, 0, stream>>>(Bd0, N0, DIM, pool_w, pool_b, sc);
  k_topk<<<1, 1024, 0, stream>>>(sc, N0, K1, idx0, val0);
  k_gather_rows<<<((K1 * DIM) + 255) / 256, 256, 0, stream>>>(Bd0, idx0, val0, K1, DIM, B0);
  {
    dim3 g((K1 + 15) / 16, (K1 + 15) / 16);
    k_adj_gather<<<g, dim3(16, 16), 0, stream>>>(A0, N0, idx0, K1, A1);
  }
  // down 1 (+residual) -> Bd1, then pool to K2
  gat(A1, K1, B0, DIM, down_w + DW, down_b + DB, down_phi + DP, DH, Bd1, DIM, B0, nullptr);
  k_pool_scores<<<(K1 + 255) / 256, 256, 0, stream>>>(Bd1, K1, DIM, pool_w + DIM, pool_b + 1, sc);
  k_topk<<<1, 1024, 0, stream>>>(sc, K1, K2, idx1, val1);
  k_gather_rows<<<((K2 * DIM) + 255) / 256, 256, 0, stream>>>(Bd1, idx1, val1, K2, DIM, B1);
  {
    dim3 g((K2 + 15) / 16, (K2 + 15) / 16);
    k_adj_gather<<<g, dim3(16, 16), 0, stream>>>(A1, K1, idx1, K2, A2);
  }
  // bottom GAT (+residual)
  gat(A2, K2, B1, DIM, bottom_w, bottom_b, bottom_phi, DH, B2, DIM, B1, nullptr);
  // up 0: scatter to K1, GAT(+residual,+skip Bd1)
  k_zero<<<((K1 * DIM) + 255) / 256, 256, 0, stream>>>(B0, K1 * DIM);
  k_scatter_rows<<<((K2 * DIM) + 255) / 256, 256, 0, stream>>>(B0, B2, idx1, K2, DIM);
  gat(A1, K1, B0, DIM, up_w, up_b, up_phi, DH, B1, DIM, B0, Bd1);
  // up 1: scatter to N0, GAT(+residual,+skip Bd0)
  k_zero<<<((N0 * DIM) + 255) / 256, 256, 0, stream>>>(B2, N0 * DIM);
  k_scatter_rows<<<((K1 * DIM) + 255) / 256, 256, 0, stream>>>(B2, B1, idx0, K1, DIM);
  gat(A0, N0, B2, DIM, up_w + DW, up_b + DB, up_phi + DP, DH, B0, DIM, B2, Bd0);
  // concat + end GAT -> first output
  k_concat<<<((N0 * 600) + 255) / 256, 256, 0, stream>>>(B0, Borg, N0, DIM, DIM, Bcat);
  gat(A0, N0, Bcat, 2 * DIM, end_w, end_b, end_phi, DHE, out, ODIM, nullptr, nullptr);
  // second output: org_X[:, :268]
  k_copy_cols<<<((N0 * ODIM) + 255) / 256, 256, 0, stream>>>(Borg, DIM, N0, ODIM,
                                                             out + (size_t)N0 * ODIM);
}
